// EuclideanCodebook_438086664506
// MI455X (gfx1250) — matrix operand order repaired and checked by reading, not compiled, not run on hardware
//
#include <hip/hip_runtime.h>
#include <hip/hip_bf16.h>

typedef __bf16 v8bf  __attribute__((ext_vector_type(8)));
typedef __bf16 v16bf __attribute__((ext_vector_type(16)));
typedef float  v8f   __attribute__((ext_vector_type(8)));

#define Nn 512
#define Mm 512
#define Cc 64
#define CS 512
#define BSTRIDE 72                  // bf16 elems/row: 36 DWORDs -> conflict-free b128 frags
#define PLANE   (Mm * BSTRIDE)      // 36864 bf16 elems = 73728 B per plane

// LDS byte offsets
#define XH_OFF 0
#define XL_OFF (1 * PLANE * 2)
#define EH_OFF (2 * PLANE * 2)
#define EL_OFF (3 * PLANE * 2)
#define XN_OFF (4 * PLANE * 2)              // 294912: xnorm f32[512]
#define EN_OFF (XN_OFF + 2048)              // enorm f32[512]
#define DM_OFF (EN_OFF + 2048)              // dmin  f32[512]
#define DI_OFF (DM_OFF + 2048)              // didx  i32[512]
#define IB_OFF (DI_OFF + 2048)              // broadcast i32[2]
#define SMEM_BYTES (IB_OFF + 16)            // 303,120 B  (< 320 KB CDNA5 WGP LDS)

// flat output layout
#define CURR_OFF  (Nn * Cc)                 // argmin_curr: 512 floats
#define PREV_OFF  (Nn * Cc + Nn)            // argmin_prev: 512*512 floats

static __device__ __forceinline__ unsigned short f2bf(float f) {
    unsigned u = __float_as_uint(f);
    return (unsigned short)((u + 0x7FFFu + ((u >> 16) & 1u)) >> 16);   // RNE
}
static __device__ __forceinline__ float bf2f(unsigned short b) {
    return __uint_as_float(((unsigned)b) << 16);
}

__global__ __launch_bounds__(256) void
codebook_bf16x2_kernel(const float* __restrict__ x,
                       const float* __restrict__ embed,
                       float* __restrict__ out) {
    extern __shared__ char smem_raw[];
    __bf16* x_hi = (__bf16*)(smem_raw + XH_OFF);
    __bf16* x_lo = (__bf16*)(smem_raw + XL_OFF);
    __bf16* e_hi = (__bf16*)(smem_raw + EH_OFF);
    __bf16* e_lo = (__bf16*)(smem_raw + EL_OFF);
    float*  xnorm = (float*)(smem_raw + XN_OFF);
    float*  enorm = (float*)(smem_raw + EN_OFF);
    float*  dminv = (float*)(smem_raw + DM_OFF);
    int*    didx  = (int*)  (smem_raw + DI_OFF);
    int*    ibc   = (int*)  (smem_raw + IB_OFF);

    const int n    = blockIdx.x;
    const int tid  = threadIdx.x;
    const int wave = tid >> 5;
    const int lane = tid & 31;
    const int half = lane >> 4;
    const int l15  = lane & 15;

    // ---- Stage x[n] and embed as split bf16 (hi+lo) planes ----
    {
        const float4* xs = (const float4*)(x + (size_t)n * (Mm * Cc));
        const float4* es = (const float4*)embed;
        for (int i = tid; i < (Mm * Cc) / 4; i += 256) {
            int g = i * 4, row = g >> 6, col = g & 63;
            float4 vx = xs[i];
            float4 ve = es[i];
            float fx[4] = {vx.x, vx.y, vx.z, vx.w};
            float fe[4] = {ve.x, ve.y, ve.z, ve.w};
            unsigned hx[4], lx[4], he[4], le[4];
            #pragma unroll
            for (int j = 0; j < 4; ++j) {
                hx[j] = f2bf(fx[j]);  lx[j] = f2bf(fx[j] - bf2f((unsigned short)hx[j]));
                he[j] = f2bf(fe[j]);  le[j] = f2bf(fe[j] - bf2f((unsigned short)he[j]));
            }
            uint2 ph = {hx[0] | (hx[1] << 16), hx[2] | (hx[3] << 16)};
            uint2 pl = {lx[0] | (lx[1] << 16), lx[2] | (lx[3] << 16)};
            uint2 qh = {he[0] | (he[1] << 16), he[2] | (he[3] << 16)};
            uint2 ql = {le[0] | (le[1] << 16), le[2] | (le[3] << 16)};
            int o = row * BSTRIDE + col;
            *(uint2*)&x_hi[o] = ph;  *(uint2*)&x_lo[o] = pl;
            *(uint2*)&e_hi[o] = qh;  *(uint2*)&e_lo[o] = ql;
        }
    }
    __syncthreads();

    // ---- Row norms from reconstructed hi+lo ----
    for (int r = tid; r < 512; r += 256) {
        float sx = 0.f, se = 0.f;
        #pragma unroll 8
        for (int k = 0; k < Cc; ++k) {
            float a = (float)x_hi[r * BSTRIDE + k] + (float)x_lo[r * BSTRIDE + k];
            float b = (float)e_hi[r * BSTRIDE + k] + (float)e_lo[r * BSTRIDE + k];
            sx = fmaf(a, a, sx);
            se = fmaf(b, b, se);
        }
        xnorm[r] = sx;
        enorm[r] = se;
    }
    __syncthreads();

    // ---- Fused GEMM (bf16x2 split via v_wmma_f32_16x16x32_bf16) + min/argmin over M ----
    for (int tt = 0; tt < 4; ++tt) {
        const int colbase = wave * 64 + tt * 16;
        const int c = colbase + l15;

        // B fragments: lane l<16 -> 16 contiguous K at kb + half*16 (contiguous k order, same as A)
        const int bo = c * BSTRIDE + half * 16;
        v16bf bh0 = *(const v16bf*)&e_hi[bo +  0];
        v16bf bh1 = *(const v16bf*)&e_hi[bo + 32];
        v16bf bl0 = *(const v16bf*)&e_lo[bo +  0];
        v16bf bl1 = *(const v16bf*)&e_lo[bo + 32];

        float runMin = 3.0e38f;
        int   runIdx = 0;

        for (int chunk = 0; chunk < 32; ++chunk) {
            const int m  = chunk * 16 + l15;
            const int ao = m * BSTRIDE + half * 16;     // A: K {0-15} / {16-31}
            v8bf ah0a = *(const v8bf*)&x_hi[ao +  0];
            v8bf ah0b = *(const v8bf*)&x_hi[ao +  8];
            v8bf ah1a = *(const v8bf*)&x_hi[ao + 32];
            v8bf ah1b = *(const v8bf*)&x_hi[ao + 40];
            v8bf al0a = *(const v8bf*)&x_lo[ao +  0];
            v8bf al0b = *(const v8bf*)&x_lo[ao +  8];
            v8bf al1a = *(const v8bf*)&x_lo[ao + 32];
            v8bf al1b = *(const v8bf*)&x_lo[ao + 40];
            v16bf Ah0 = __builtin_shufflevector(ah0a, ah0b, 0,1,2,3,4,5,6,7,8,9,10,11,12,13,14,15);
            v16bf Ah1 = __builtin_shufflevector(ah1a, ah1b, 0,1,2,3,4,5,6,7,8,9,10,11,12,13,14,15);
            v16bf Al0 = __builtin_shufflevector(al0a, al0b, 0,1,2,3,4,5,6,7,8,9,10,11,12,13,14,15);
            v16bf Al1 = __builtin_shufflevector(al1a, al1b, 0,1,2,3,4,5,6,7,8,9,10,11,12,13,14,15);

            v8f acc = {0.f, 0.f, 0.f, 0.f, 0.f, 0.f, 0.f, 0.f};
            acc = __builtin_amdgcn_wmma_f32_16x16x32_bf16(false, Ah0, false, bh0, (short)0, acc, false, false);
            acc = __builtin_amdgcn_wmma_f32_16x16x32_bf16(false, Ah1, false, bh1, (short)0, acc, false, false);
            acc = __builtin_amdgcn_wmma_f32_16x16x32_bf16(false, Ah0, false, bl0, (short)0, acc, false, false);
            acc = __builtin_amdgcn_wmma_f32_16x16x32_bf16(false, Ah1, false, bl1, (short)0, acc, false, false);
            acc = __builtin_amdgcn_wmma_f32_16x16x32_bf16(false, Al0, false, bh0, (short)0, acc, false, false);
            acc = __builtin_amdgcn_wmma_f32_16x16x32_bf16(false, Al1, false, bh1, (short)0, acc, false, false);

            const int rowbase = chunk * 16 + (half << 3);
            #pragma unroll
            for (int r = 0; r < 8; ++r) {
                float val = fmaf(acc[r], -2.0f, xnorm[rowbase + r]);
                if (val < runMin) { runMin = val; runIdx = rowbase + r; }
            }
        }

        // lanes c and c+16 hold the same column: combine, first-index on ties
        float oVal = __shfl_xor(runMin, 16, 32);
        int   oIdx = __shfl_xor(runIdx, 16, 32);
        if (oVal < runMin || (oVal == runMin && oIdx < runIdx)) {
            runMin = oVal; runIdx = oIdx;
        }
        if (half == 0) {
            dminv[c] = runMin + enorm[c];
            didx[c]  = runIdx;
            out[PREV_OFF + (size_t)n * CS + c] = (float)runIdx;
        }
    }
    __syncthreads();

    // ---- argmin over the 512 codes (wave 0) ----
    if (wave == 0) {
        float best = 3.0e38f; int bc = 0;
        for (int i = 0; i < 16; ++i) {
            int c = lane + 32 * i;
            float v = dminv[c];
            if (v < best) { best = v; bc = c; }
        }
        #pragma unroll
        for (int off = 16; off >= 1; off >>= 1) {
            float ov = __shfl_xor(best, off, 32);
            int   oc = __shfl_xor(bc,   off, 32);
            if (ov < best || (ov == best && oc < bc)) { best = ov; bc = oc; }
        }
        if (lane == 0) {
            ibc[0] = bc;
            ibc[1] = didx[bc];
            out[CURR_OFF + n] = (float)bc;
        }
    }
    __syncthreads();

    // ---- residual = x[n, m*, :] - embed[c*, :]  (exact f32, from global) ----
    const int cstar = ibc[0];
    const int mstar = ibc[1];
    if (tid < Cc)
        out[(size_t)n * Cc + tid] =
            x[(size_t)n * (Mm * Cc) + (size_t)mstar * Cc + tid] -
            embed[(size_t)cstar * Cc + tid];
}

extern "C" void kernel_launch(void* const* d_in, const int* in_sizes, int n_in,
                              void* d_out, int out_size, void* d_ws, size_t ws_size,
                              hipStream_t stream) {
    (void)in_sizes; (void)n_in; (void)d_ws; (void)ws_size; (void)out_size;
    const float* x     = (const float*)d_in[0];
    const float* embed = (const float*)d_in[1];
    float* out = (float*)d_out;

    hipFuncSetAttribute((const void*)codebook_bf16x2_kernel,
                        hipFuncAttributeMaxDynamicSharedMemorySize, SMEM_BYTES);
    codebook_bf16x2_kernel<<<dim3(Nn), dim3(256), SMEM_BYTES, stream>>>(x, embed, out);
}
